// C3_layer_35321811042899
// MI455X (gfx1250) — compile-verified, matrix operand order checked by reading, not run on hardware
//
#include <hip/hip_runtime.h>

typedef __attribute__((ext_vector_type(16))) __bf16 v16bf;
typedef __attribute__((ext_vector_type(8)))  float  v8f;
typedef __attribute__((ext_vector_type(4)))  float  v4f;

#define TILE_W 128          // output columns per block (8 waves * 16)
#define LDS_W  132          // TILE_W + 4 halo
#define OUT_W  508
#define IN_W   512
#define HALF_PLANE (3 * 5 * LDS_W)   // 3 channel planes (hi-lane base shift)

__constant__ int c_CH3[6][3] = {{0,1,2},{1,2,3},{2,3,4},{3,4,5},{0,4,5},{0,1,5}};
__constant__ int c_CH4[9][4] = {{0,1,2,3},{1,2,3,4},{2,3,4,5},{0,3,4,5},{0,1,4,5},
                                {0,1,2,5},{0,1,3,4},{1,2,4,5},{0,2,3,5}};

// ---------------------------------------------------------------------------
// K-slot mapping (shared by A-pack and B-build):
//   physical slot = (k32, e, hi) with hi = (lane >= 16)
//   t = k32*16 + e  (0..79)
//   t < 75 : logical tap = channel (t/25 + 3*hi), ky = (t%25)/5, kx = t%5
//   t >= 75: zero pad slot
// Every B-gather offset is a compile-time constant; `hi` only shifts the LDS
// base by 3 channel planes.
// ---------------------------------------------------------------------------

// Prep: pack weights into the per-lane A-fragment layout of
// v_wmma_f32_16x16x32_bf16 (lane = out-channel M, element e = K slot) + bias.
__global__ void prep_pack_kernel(const float* __restrict__ w3, const float* __restrict__ b3,
                                 const float* __restrict__ w4, const float* __restrict__ b4,
                                 const float* __restrict__ w6, const float* __restrict__ b6,
                                 unsigned short* __restrict__ packedW_u16,
                                 float* __restrict__ biasOut) {
    __bf16* packedW = (__bf16*)packedW_u16;
    for (int idx = threadIdx.x; idx < 5 * 32 * 16; idx += blockDim.x) {
        int k32  = idx >> 9;            // idx / 512
        int lane = (idx >> 4) & 31;
        int e    = idx & 15;
        int hi   = (lane >= 16) ? 1 : 0;
        int t    = k32 * 16 + e;
        int oc   = lane & 15;
        float v = 0.0f;
        if (t < 75) {
            int c = t / 25 + 3 * hi;    // input channel 0..5
            int r = t % 25;             // ky*5 + kx
            if (oc < 6) {
                for (int j = 0; j < 3; ++j)
                    if (c_CH3[oc][j] == c) v = w3[(oc * 3 + j) * 25 + r];
            } else if (oc < 15) {
                int o = oc - 6;
                for (int j = 0; j < 4; ++j)
                    if (c_CH4[o][j] == c) v = w4[(o * 4 + j) * 25 + r];
            } else {
                v = w6[c * 25 + r];
            }
        }
        packedW[idx] = (__bf16)v;
    }
    if (threadIdx.x < 16) {
        int ch = threadIdx.x;
        biasOut[ch] = (ch < 6) ? b3[ch] : (ch < 15 ? b4[ch - 6] : b6[0]);
    }
}

// ---------------------------------------------------------------------------
// Main: implicit-GEMM conv via v_wmma_f32_16x16x32_bf16.
//   A (16x160)  = packed weights (M = out channel), 5 resident fragments
//   B (160x16)  = input patch columns (N = 16 consecutive output x) gathered
//                 from LDS with compile-time immediate offsets
//   C           = bias (folded into accumulator init)
//   Fill        = vectorized b128 global->LDS staging (990 float4 tiles)
//   D store     : per acc VGPR, 16 contiguous floats of one channel row
//                 -> coalesced nontemporal (streaming) stores.
// Grid: (ceil(508/128), 508, 32), block 256 = 8 wave32.
// ---------------------------------------------------------------------------
__global__ __launch_bounds__(256)
void conv5x5_wmma_kernel(const float* __restrict__ x,
                         const unsigned short* __restrict__ packedW_u16,
                         const float* __restrict__ bias,
                         float* __restrict__ out) {
    __shared__ float lds[6 * 5 * LDS_W];

    const int x0  = blockIdx.x * TILE_W;   // first output column (mult. of 128)
    const int y   = blockIdx.y;            // output row
    const int b   = blockIdx.z;            // batch
    const int tid = threadIdx.x;

    // Stage input halo as 990 aligned float4 tiles: 30 (c,ky) planes x 33 tiles.
    // Right-edge clamp (gx==512, only bx==3 last tile) is whole-vector.
    for (int i = tid; i < 6 * 5 * 33; i += 256) {
        int p    = i / 33;                  // c*5 + ky
        int col4 = i - p * 33;
        int c    = p / 5;
        int ky   = p - c * 5;
        int gx   = x0 + col4 * 4;
        v4f v = {0.0f, 0.0f, 0.0f, 0.0f};
        if (gx < IN_W)                      // gx<=508 -> gx+3<=511 in range
            v = *(const v4f*)&x[((b * 6 + c) * (size_t)IN_W + (size_t)(y + ky)) * IN_W + gx];
        *(v4f*)&lds[p * LDS_W + col4 * 4] = v;
    }
    __syncthreads();

    const int lane = tid & 31;
    const int wave = tid >> 5;
    const int hi   = (lane >= 16) ? 1 : 0;
    const int xm   = wave * 16 + (lane & 15);      // column within tile (0..127)

    // Per-lane gather base: hi lanes read channels 3..5.
    const float* __restrict__ gbase = lds + hi * HALF_PLANE + xm;

    // Resident A fragments (weights), already in WMMA lane layout: 32B each.
    v16bf afrag[5];
#pragma unroll
    for (int k32 = 0; k32 < 5; ++k32)
        afrag[k32] = *(const v16bf*)(packedW_u16 + (k32 * 32 + lane) * 16);

    // Accumulator starts at bias (WMMA C-operand does the bias add for free).
    v8f acc;
#pragma unroll
    for (int v = 0; v < 8; ++v)
        acc[v] = bias[v + hi * 8];

#pragma unroll
    for (int k32 = 0; k32 < 5; ++k32) {
        v16bf bfrag;
#pragma unroll
        for (int e = 0; e < 16; ++e) {
            const int t = k32 * 16 + e;            // compile-time
            if (t < 75) {
                const int c  = t / 25;             // 0..2 (hi handled by gbase)
                const int r  = t - c * 25;
                const int ky = r / 5;
                const int kx = r - ky * 5;
                // compile-time immediate ds offset; single shared addr VGPR
                bfrag[e] = (__bf16)gbase[(c * 5 + ky) * LDS_W + kx];
            } else {
                bfrag[e] = (__bf16)0.0f;           // K padding slots
            }
        }
        // D = A*B + C ; EXEC all-1s (no divergence anywhere above).
        acc = __builtin_amdgcn_wmma_f32_16x16x32_bf16(
                  false, afrag[k32], false, bfrag,
                  (short)0, acc, false, false);
    }

    // Store: per VGPR v, 16 contiguous x positions of channel (v + 8*hi).
    const int xout = x0 + wave * 16 + (lane & 15);
    if (xout < OUT_W) {
#pragma unroll
        for (int v = 0; v < 8; ++v) {
            int ch = v + hi * 8;
            __builtin_nontemporal_store(
                acc[v], &out[((b * 16 + ch) * (size_t)OUT_W + y) * OUT_W + xout]);
        }
    }
}

extern "C" void kernel_launch(void* const* d_in, const int* in_sizes, int n_in,
                              void* d_out, int out_size, void* d_ws, size_t ws_size,
                              hipStream_t stream) {
    const float* x  = (const float*)d_in[0];
    const float* w3 = (const float*)d_in[1];
    const float* b3 = (const float*)d_in[2];
    const float* w4 = (const float*)d_in[3];
    const float* b4 = (const float*)d_in[4];
    const float* w6 = (const float*)d_in[5];
    const float* b6 = (const float*)d_in[6];

    unsigned short* packedW = (unsigned short*)d_ws;                   // 5120 B
    float*          bias    = (float*)((char*)d_ws + 5 * 32 * 16 * 2); // 16 f32
    float*          out     = (float*)d_out;

    prep_pack_kernel<<<1, 256, 0, stream>>>(w3, b3, w4, b4, w6, b6, packedW, bias);

    dim3 grid((OUT_W + TILE_W - 1) / TILE_W, OUT_W, 32);
    conv5x5_wmma_kernel<<<grid, 256, 0, stream>>>(x, packedW, bias, out);
}
